// HGAT90_32375463477978
// MI455X (gfx1250) — matrix operand order repaired and checked by reading, not compiled, not run on hardware
//
#include <hip/hip_runtime.h>
#include <hip/hip_bf16.h>
#include <stdint.h>

#define H_ 8
#define C_ 128
#define HC_ 1024
#define NN_ 8192
#define E_ 98304
#define ETOT_ (E_ + NN_)
#define G_ 64
#define IN_ 128
#define NTYPE_ 32
#define NFAM_ 64
#define TE_ 16
#define EPS_ 1e-5f
#define SLOPE_ 0.2f

typedef __bf16 bf16;
typedef bf16  v16bf __attribute__((ext_vector_type(16)));
typedef float v8f   __attribute__((ext_vector_type(8)));
typedef unsigned int u32x4 __attribute__((ext_vector_type(4)));

union FragU { v16bf v; u32x4 q[2]; };

// ---------------- helpers ----------------
__device__ __forceinline__ unsigned ordF(float f) {
  unsigned u = __float_as_uint(f);
  return (u & 0x80000000u) ? ~u : (u | 0x80000000u);
}
__device__ __forceinline__ float deordF(unsigned u) {
  unsigned b = (u & 0x80000000u) ? (u & 0x7FFFFFFFu) : ~u;
  return __uint_as_float(b);
}

// async DMA: global -> LDS, 16 bytes, ASYNCcnt-tracked.
// LDS offset = low 32 bits of the generic pointer.
__device__ __forceinline__ void async_g2l_b128(const void* gptr, void* lptr) {
  uint32_t la = (uint32_t)(uintptr_t)lptr;
  uint64_t ga = (uint64_t)(uintptr_t)gptr;
  asm volatile("global_load_async_to_lds_b128 %0, %1, off" :: "v"(la), "v"(ga) : "memory");
}
__device__ __forceinline__ void wait_async0() {
  asm volatile("s_wait_asynccnt 0x0" ::: "memory");
}

// ---------------- generic fill / convert ----------------
__global__ void fill0_kernel(unsigned* __restrict__ p, int n) {
  int i = blockIdx.x * blockDim.x + threadIdx.x;
  if (i < n) p[i] = 0u;
}

__global__ void f2bf_kernel(const float* __restrict__ a, bf16* __restrict__ o, int n) {
  int i = blockIdx.x * blockDim.x + threadIdx.x;
  if (i < n) o[i] = (bf16)a[i];
}

// transpose + convert weights: wt[n][k] = (bf16)w[k][n]; LDS-tiled so both sides coalesce
__global__ __launch_bounds__(256)
void wtrans_kernel(const float* __restrict__ w, bf16* __restrict__ wt, int K, int Nd) {
  __shared__ float tile[32][33];
  int bx = blockIdx.x, by = blockIdx.y;          // bx: n-tile, by: k-tile
  int tx = threadIdx.x & 31, ty = threadIdx.x >> 5;  // 32 x 8
  #pragma unroll
  for (int j = 0; j < 4; ++j)
    tile[ty + j * 8][tx] = w[(size_t)(by * 32 + ty + j * 8) * Nd + bx * 32 + tx];
  __syncthreads();
  #pragma unroll
  for (int j = 0; j < 4; ++j)
    wt[(size_t)(bx * 32 + ty + j * 8) * K + by * 32 + tx] = (bf16)tile[tx][ty + j * 8];
}

// ---------------- bf16 WMMA GEMM: C = A[MxK] * Bt[NxK]^T + bias ----------------
// Block tile 256x128, 8 waves of 64x64 (4x4 WMMA tiles each).
// Double-buffered LDS staged entirely with global_load_async_to_lds_b128.
#define BM 256
#define BN 128
#define BK 32
#define LDSS 40   // padded bf16 row stride (80 B)

__global__ __launch_bounds__(256)
void gemm_bf16_kernel(const bf16* __restrict__ A, const bf16* __restrict__ Bt,
                      const float* __restrict__ bias, float* __restrict__ Cout,
                      int M, int Nd, int K) {
  (void)M;
  __shared__ bf16 smA[2][BM * LDSS];   // 2 x 20 KB
  __shared__ bf16 smB[2][BN * LDSS];   // 2 x 10 KB

  const int tid  = threadIdx.x;
  const int wid  = tid >> 5;
  const int lane = tid & 31;
  const int wm   = wid & 3;          // wave row (4 in M)
  const int wn   = wid >> 2;         // wave col (2 in N)
  const int m0   = blockIdx.x * BM;
  const int n0   = blockIdx.y * BN;
  const int laneM = lane & 15;
  const int khalf = lane >> 4;

  // A tile: 256 rows x 32 bf16 = 1024 x 16B chunks (4/thread);
  // B tile: 128 rows x 32 bf16 =  512 x 16B chunks (2/thread). No loop guards.
  auto stageA1 = [&](int buf, int k0, int c) {
    int row = c >> 2, seg = c & 3;
    async_g2l_b128(A + (size_t)(m0 + row) * K + k0 + seg * 8,
                   &smA[buf][row * LDSS + seg * 8]);
  };
  auto stageB1 = [&](int buf, int k0, int c) {
    int row = c >> 2, seg = c & 3;
    async_g2l_b128(Bt + (size_t)(n0 + row) * K + k0 + seg * 8,
                   &smB[buf][row * LDSS + seg * 8]);
  };
  auto stage = [&](int buf, int k0) {
    stageA1(buf, k0, tid);
    stageA1(buf, k0, tid + 256);
    stageA1(buf, k0, tid + 512);
    stageA1(buf, k0, tid + 768);
    stageB1(buf, k0, tid);
    stageB1(buf, k0, tid + 256);
  };

  v8f c[4][4] = {};

  const int ktiles = K / BK;
  stage(0, 0);
  wait_async0();
  __syncthreads();

  for (int kt = 0; kt < ktiles; ++kt) {
    const int buf = kt & 1;
    if (kt + 1 < ktiles)              // overlap next-tile DMA with WMMA
      stage(buf ^ 1, (kt + 1) * BK);

    FragU a[4], b[4];
    #pragma unroll
    for (int mi = 0; mi < 4; ++mi) {
      int m = wm * 64 + mi * 16 + laneM;
      const u32x4* p = (const u32x4*)(&smA[buf][m * LDSS + 8 * khalf]);
      a[mi].q[0] = p[0];   // k = 8*khalf .. +7
      a[mi].q[1] = p[2];   // k = 16+8*khalf .. +7
    }
    #pragma unroll
    for (int ni = 0; ni < 4; ++ni) {
      int n = wn * 64 + ni * 16 + laneM;
      const u32x4* p = (const u32x4*)(&smB[buf][n * LDSS + 8 * khalf]);
      b[ni].q[0] = p[0];   // k = 8*khalf .. +7
      b[ni].q[1] = p[2];   // k = 16+8*khalf .. +7
    }
    #pragma unroll
    for (int mi = 0; mi < 4; ++mi)
      #pragma unroll
      for (int ni = 0; ni < 4; ++ni)
        c[mi][ni] = __builtin_amdgcn_wmma_f32_16x16x32_bf16(
            false, a[mi].v, false, b[ni].v, (short)0, c[mi][ni], false, false);

    wait_async0();
    __syncthreads();
  }

  // epilogue: bias add + f32 store
  #pragma unroll
  for (int ni = 0; ni < 4; ++ni) {
    int col = n0 + wn * 64 + ni * 16 + laneM;
    float bv = bias[col];
    #pragma unroll
    for (int mi = 0; mi < 4; ++mi) {
      int rbase = m0 + wm * 64 + mi * 16 + 8 * khalf;
      #pragma unroll
      for (int r = 0; r < 8; ++r)
        Cout[(size_t)(rbase + r) * Nd + col] = c[mi][ni][r] + bv;
    }
  }
}

// ---------------- edge phase ----------------
// block = edge, wave = head, lane covers 4 channels
__global__ __launch_bounds__(256)
void logits_kernel(const float* __restrict__ xl, const float* __restrict__ xr,
                   const float* __restrict__ att, const int* __restrict__ esrc,
                   const int* __restrict__ edst, float* __restrict__ lg,
                   unsigned* __restrict__ lmaxU) {
  int e = blockIdx.x;
  int hh = threadIdx.x >> 5, lane = threadIdx.x & 31;
  int src = (e < E_) ? esrc[e] : (e - E_);
  int dst = (e < E_) ? edst[e] : (e - E_);
  const float4 a = *(const float4*)(att + hh * C_ + lane * 4);
  const float4 l = *(const float4*)(xl + (size_t)src * HC_ + hh * C_ + lane * 4);
  const float4 r = *(const float4*)(xr + (size_t)dst * HC_ + hh * C_ + lane * 4);
  float s = 0.f, m;
  m = l.x + r.x; m = (m > 0.f) ? m : SLOPE_ * m; s += m * a.x;
  m = l.y + r.y; m = (m > 0.f) ? m : SLOPE_ * m; s += m * a.y;
  m = l.z + r.z; m = (m > 0.f) ? m : SLOPE_ * m; s += m * a.z;
  m = l.w + r.w; m = (m > 0.f) ? m : SLOPE_ * m; s += m * a.w;
  #pragma unroll
  for (int o = 16; o; o >>= 1) s += __shfl_xor(s, o, 32);
  if (lane == 0) {
    lg[(size_t)e * H_ + hh] = s;
    atomicMax(&lmaxU[(size_t)dst * H_ + hh], ordF(s));
  }
}

__global__ void exp_kernel(float* __restrict__ lg, const unsigned* __restrict__ lmaxU,
                           float* __restrict__ denom, const int* __restrict__ edst) {
  int idx = blockIdx.x * blockDim.x + threadIdx.x;
  if (idx >= ETOT_ * H_) return;
  int e = idx >> 3, hh = idx & 7;
  int dst = (e < E_) ? edst[e] : (e - E_);
  float lm = deordF(lmaxU[dst * H_ + hh]);
  float ex = __expf(lg[idx] - lm);
  lg[idx] = ex;
  atomicAdd(&denom[dst * H_ + hh], ex);
}

__global__ __launch_bounds__(256)
void scatter_kernel(const float* __restrict__ xl, const float* __restrict__ ealpha,
                    const float* __restrict__ denom, const int* __restrict__ esrc,
                    const int* __restrict__ edst, float* __restrict__ acc) {
  int e = blockIdx.x;
  int hh = threadIdx.x >> 5, lane = threadIdx.x & 31;
  int src = (e < E_) ? esrc[e] : (e - E_);
  int dst = (e < E_) ? edst[e] : (e - E_);
  float alpha = ealpha[(size_t)e * H_ + hh] / denom[(size_t)dst * H_ + hh];
  const float4 v = *(const float4*)(xl + (size_t)src * HC_ + hh * C_ + lane * 4);
  float* d = acc + (size_t)dst * HC_ + hh * C_ + lane * 4;
  atomicAdd(d + 0, v.x * alpha);
  atomicAdd(d + 1, v.y * alpha);
  atomicAdd(d + 2, v.z * alpha);
  atomicAdd(d + 3, v.w * alpha);
}

// bias + ReLU + LayerNorm (+residual) + write f32 h and bf16 copy
__global__ __launch_bounds__(256)
void ln_kernel(const float* __restrict__ acc, const float* __restrict__ bias,
               const float* __restrict__ gam, const float* __restrict__ bet,
               const float* __restrict__ hprev, float* __restrict__ h,
               bf16* __restrict__ hb) {
  int n = blockIdx.x, t = threadIdx.x;
  int c0 = t * 4;
  const float4 a4 = *(const float4*)(acc + (size_t)n * HC_ + c0);
  const float4 b4 = *(const float4*)(bias + c0);
  float z[4];
  z[0] = fmaxf(a4.x + b4.x, 0.f);
  z[1] = fmaxf(a4.y + b4.y, 0.f);
  z[2] = fmaxf(a4.z + b4.z, 0.f);
  z[3] = fmaxf(a4.w + b4.w, 0.f);
  float s = z[0] + z[1] + z[2] + z[3];
  float s2 = z[0]*z[0] + z[1]*z[1] + z[2]*z[2] + z[3]*z[3];
  #pragma unroll
  for (int o = 16; o; o >>= 1) { s += __shfl_xor(s, o, 32); s2 += __shfl_xor(s2, o, 32); }
  __shared__ float rs[8], rs2[8];
  int wid = t >> 5, lane = t & 31;
  if (lane == 0) { rs[wid] = s; rs2[wid] = s2; }
  __syncthreads();
  float S = 0.f, S2 = 0.f;
  #pragma unroll
  for (int i = 0; i < 8; ++i) { S += rs[i]; S2 += rs2[i]; }
  float mean = S * (1.f / HC_);
  float var  = S2 * (1.f / HC_) - mean * mean;
  float inv  = rsqrtf(var + EPS_);
  #pragma unroll
  for (int j = 0; j < 4; ++j) {
    int c = c0 + j;
    float o = (z[j] - mean) * inv * gam[c] + bet[c];
    if (hprev) o += hprev[(size_t)n * HC_ + c];
    h[(size_t)n * HC_ + c] = o;
    hb[(size_t)n * HC_ + c] = (bf16)o;
  }
}

// ---------------- pooling + heads ----------------
__global__ __launch_bounds__(256)
void pool_kernel(const float* __restrict__ h, const int* __restrict__ batch,
                 float* __restrict__ gfeat, float* __restrict__ cnt) {
  int n = blockIdx.x, t = threadIdx.x;
  int g = batch[n];
  int c0 = t * 4;
  const float4 v = *(const float4*)(h + (size_t)n * HC_ + c0);
  float* d = gfeat + (size_t)g * (HC_ + TE_) + c0;
  atomicAdd(d + 0, v.x);
  atomicAdd(d + 1, v.y);
  atomicAdd(d + 2, v.z);
  atomicAdd(d + 3, v.w);
  if (t == 0) atomicAdd(&cnt[g], 1.f);
}

__global__ void poolfin_kernel(float* __restrict__ gfeat, const float* __restrict__ cnt,
                               const float* __restrict__ Temb, const int* __restrict__ y_type) {
  int idx = blockIdx.x * blockDim.x + threadIdx.x;
  if (idx >= G_ * (HC_ + TE_)) return;
  int g = idx / (HC_ + TE_);
  int c = idx - g * (HC_ + TE_);
  if (c < HC_) gfeat[idx] = gfeat[idx] / fmaxf(cnt[g], 1.f);
  else         gfeat[idx] = Temb[y_type[g] * TE_ + (c - HC_)];
}

__global__ void head_kernel(const float* __restrict__ gfeat,
                            const float* __restrict__ W_fam, const float* __restrict__ b_fam,
                            const float* __restrict__ W_type, const float* __restrict__ b_type,
                            float* __restrict__ out) {
  int idx = blockIdx.x * blockDim.x + threadIdx.x;
  const int D = HC_ + TE_;
  if (idx < G_ * NFAM_) {
    int g = idx / NFAM_, o = idx - g * NFAM_;
    float s = b_fam[o];
    for (int k = 0; k < D; ++k) s += gfeat[g * D + k] * W_fam[k * NFAM_ + o];
    out[idx] = s;
  } else if (idx < G_ * NFAM_ + G_ * NTYPE_) {
    int j = idx - G_ * NFAM_;
    int g = j / NTYPE_, o = j - g * NTYPE_;
    float s = b_type[o];
    for (int k = 0; k < D; ++k) s += gfeat[g * D + k] * W_type[k * NTYPE_ + o];
    out[G_ * NFAM_ + j] = s;
  }
}

// ---------------- host orchestration ----------------
extern "C" void kernel_launch(void* const* d_in, const int* in_sizes, int n_in,
                              void* d_out, int out_size, void* d_ws, size_t ws_size,
                              hipStream_t stream) {
  (void)in_sizes; (void)n_in; (void)out_size; (void)ws_size;
  const float* x      = (const float*)d_in[0];
  const int*   edge   = (const int*)d_in[1];
  const int*   batch  = (const int*)d_in[2];
  const int*   y_type = (const int*)d_in[3];
  const float* Wl0 = (const float*)d_in[4];
  const float* bl0 = (const float*)d_in[5];
  const float* Wr0 = (const float*)d_in[6];
  const float* br0 = (const float*)d_in[7];
  const float* att0 = (const float*)d_in[8];
  const float* b0   = (const float*)d_in[9];
  const float* Wl = (const float*)d_in[10];
  const float* bl = (const float*)d_in[11];
  const float* Wr = (const float*)d_in[12];
  const float* br = (const float*)d_in[13];
  const float* att = (const float*)d_in[14];
  const float* bb  = (const float*)d_in[15];
  const float* ln_g = (const float*)d_in[16];
  const float* ln_b = (const float*)d_in[17];
  const float* Temb = (const float*)d_in[18];
  const float* W_fam = (const float*)d_in[19];
  const float* b_fam = (const float*)d_in[20];
  const float* W_type = (const float*)d_in[21];
  const float* b_type = (const float*)d_in[22];

  const int* esrc = edge;
  const int* edst = edge + E_;

  // carve workspace
  uint8_t* wp = (uint8_t*)d_ws;
  auto carve = [&](size_t bytes) -> void* {
    void* p = wp; wp += (bytes + 255) & ~(size_t)255; return p;
  };
  float*    h      = (float*)carve((size_t)NN_ * HC_ * 4);
  float*    xl     = (float*)carve((size_t)NN_ * HC_ * 4);
  float*    xr     = (float*)carve((size_t)NN_ * HC_ * 4);
  float*    acc    = (float*)carve((size_t)NN_ * HC_ * 4);
  bf16*     hb     = (bf16*)carve((size_t)NN_ * HC_ * 2);
  bf16*     wlt    = (bf16*)carve((size_t)HC_ * HC_ * 2);   // transposed bf16 weights
  bf16*     wrt    = (bf16*)carve((size_t)HC_ * HC_ * 2);
  float*    lg     = (float*)carve((size_t)ETOT_ * H_ * 4);
  unsigned* lmaxU  = (unsigned*)carve((size_t)NN_ * H_ * 4);
  float*    denom  = (float*)carve((size_t)NN_ * H_ * 4);
  float*    gfeat  = (float*)carve((size_t)G_ * (HC_ + TE_) * 4);
  float*    cnt    = (float*)carve((size_t)G_ * 4);

  auto zero = [&](void* p, int n_u32) {
    fill0_kernel<<<(n_u32 + 255) / 256, 256, 0, stream>>>((unsigned*)p, n_u32);
  };

  auto run_layer = [&](const bf16* inb, int K, const float* Wlf, const float* blf,
                       const float* Wrf, const float* brf, const float* attf,
                       const float* biasf, const float* gam, const float* bet,
                       const float* hprev) {
    dim3 tg(HC_ / 32, K / 32);
    wtrans_kernel<<<tg, 256, 0, stream>>>(Wlf, wlt, K, HC_);
    wtrans_kernel<<<tg, 256, 0, stream>>>(Wrf, wrt, K, HC_);
    dim3 gg(NN_ / BM, HC_ / BN);
    gemm_bf16_kernel<<<gg, 256, 0, stream>>>(inb, wlt, blf, xl, NN_, HC_, K);
    gemm_bf16_kernel<<<gg, 256, 0, stream>>>(inb, wrt, brf, xr, NN_, HC_, K);
    zero(lmaxU, NN_ * H_);
    zero(denom, NN_ * H_);
    zero(acc, NN_ * HC_);
    logits_kernel<<<ETOT_, 256, 0, stream>>>(xl, xr, attf, esrc, edst, lg, lmaxU);
    exp_kernel<<<(ETOT_ * H_ + 255) / 256, 256, 0, stream>>>(lg, lmaxU, denom, edst);
    scatter_kernel<<<ETOT_, 256, 0, stream>>>(xl, lg, denom, esrc, edst, acc);
    ln_kernel<<<NN_, 256, 0, stream>>>(acc, biasf, gam, bet, hprev, h, hb);
  };

  // layer 0 (input x -> bf16 into hb; consumed by GEMMs before ln overwrites hb)
  f2bf_kernel<<<(NN_ * IN_ + 255) / 256, 256, 0, stream>>>(x, hb, NN_ * IN_);
  run_layer(hb, IN_, Wl0, bl0, Wr0, br0, att0, b0, ln_g, ln_b, nullptr);
  // layers 1..3 with residual
  for (int i = 0; i < 3; ++i) {
    run_layer(hb, HC_,
              Wl + (size_t)i * HC_ * HC_, bl + (size_t)i * HC_,
              Wr + (size_t)i * HC_ * HC_, br + (size_t)i * HC_,
              att + (size_t)i * H_ * C_, bb + (size_t)i * HC_,
              ln_g + (size_t)(i + 1) * HC_, ln_b + (size_t)(i + 1) * HC_, h);
  }

  // global mean pool + type embedding + heads
  zero(gfeat, G_ * (HC_ + TE_));
  zero(cnt, G_);
  pool_kernel<<<NN_, 256, 0, stream>>>(h, batch, gfeat, cnt);
  poolfin_kernel<<<(G_ * (HC_ + TE_) + 255) / 256, 256, 0, stream>>>(gfeat, cnt, Temb, y_type);
  head_kernel<<<(G_ * (NFAM_ + NTYPE_) + 255) / 256, 256, 0, stream>>>(
      gfeat, W_fam, b_fam, W_type, b_type, (float*)d_out);
}
